// SAGE_clf_30288109371889
// MI455X (gfx1250) — matrix operand order repaired and checked by reading, not compiled, not run on hardware
//
#include <hip/hip_runtime.h>
#include <hip/hip_bf16.h>

// ---------------------------------------------------------------------------
// Problem constants (from the reference)
// ---------------------------------------------------------------------------
#define Bb      64
#define CHn     2048
#define HW2     196          // 14*14
#define Cn      1000         // graph nodes
#define Cpad    1024         // padded node rows (63*16 = 1008 computed)
#define En      40000        // edges
#define DIN     300
#define DINp    320          // padded K for layer 1
#define DH1     1024
#define DH2     2048

typedef __attribute__((ext_vector_type(16))) _Float16 v16h;
typedef __attribute__((ext_vector_type(8)))  float    v8f;

union Frag { v16h v; uint4 q[2]; };

// A fragment: 16x32 f16, row-major source A[row][ldk].
// Lane layout (ISA 7.12.2): lanes 0-15 hold K {0..7,16..23}, lanes 16-31 hold
// K {8..15,24..31}; pairs per VGPR -> two aligned 16B loads.
__device__ __forceinline__ v16h load_frag_a(const _Float16* __restrict__ A,
                                            int row, int k0, int ldk, int hi) {
    const uint4* p = (const uint4*)(A + (size_t)row * ldk + k0 + hi * 8);
    Frag f;
    f.q[0] = p[0];   // K = k0 + hi*8      .. +7
    f.q[1] = p[2];   // K = k0 + 16 + hi*8 .. +7
    return f.v;
}

// B fragment: 32x16 f16 from transposed storage Bt[col][ldk] (K contiguous).
// Lanes 0-15 hold K {0..7,16..23} of column lane%16; lanes 16-31 hold K {8..15,24..31}.
__device__ __forceinline__ v16h load_frag_b(const _Float16* __restrict__ Bt,
                                            int col, int k0, int ldk, int hi) {
    const uint4* p = (const uint4*)(Bt + (size_t)col * ldk + k0 + hi * 8);
    Frag f;
    f.q[0] = p[0];
    f.q[1] = p[2];
    return f.v;
}

// ---------------------------------------------------------------------------
// WMMA GEMM: D = A1*B1t' (+ A2*B2t') (+ bias), optional leaky-ReLU.
// One wave computes a 16x32 tile (two 16x16 column tiles sharing A fragments)
// with 2 (single) / 4 (dual) independent WMMA accumulator chains.
// ---------------------------------------------------------------------------
template <bool DUAL>
__global__ void __launch_bounds__(32)
gemm16x32_wmma_kernel(const _Float16* __restrict__ A1, const _Float16* __restrict__ B1,
                      const _Float16* __restrict__ A2, const _Float16* __restrict__ B2,
                      const float* __restrict__ bias,
                      float* __restrict__ D32, _Float16* __restrict__ D16,
                      int ldk, int K, int ldd, int mValid, int nValid, int doLeaky) {
    const int lane = threadIdx.x;
    const int li = lane & 15;
    const int hi = lane >> 4;
    const int tm = blockIdx.y, tn = blockIdx.x;
    const int row  = tm * 16 + li;
    const int col0 = tn * 32 + li;
    const int col1 = col0 + 16;

    v8f acc0 = {}, acc1 = {}, acc2 = {}, acc3 = {};
    for (int k0 = 0; k0 < K; k0 += 32) {
        v16h a1  = load_frag_a(A1, row, k0, ldk, hi);
        v16h b10 = load_frag_b(B1, col0, k0, ldk, hi);
        v16h b11 = load_frag_b(B1, col1, k0, ldk, hi);
        acc0 = __builtin_amdgcn_wmma_f32_16x16x32_f16(false, a1, false, b10,
                                                      (short)0, acc0, false, false);
        acc1 = __builtin_amdgcn_wmma_f32_16x16x32_f16(false, a1, false, b11,
                                                      (short)0, acc1, false, false);
        if constexpr (DUAL) {
            v16h a2  = load_frag_a(A2, row, k0, ldk, hi);
            v16h b20 = load_frag_b(B2, col0, k0, ldk, hi);
            v16h b21 = load_frag_b(B2, col1, k0, ldk, hi);
            acc2 = __builtin_amdgcn_wmma_f32_16x16x32_f16(false, a2, false, b20,
                                                          (short)0, acc2, false, false);
            acc3 = __builtin_amdgcn_wmma_f32_16x16x32_f16(false, a2, false, b21,
                                                          (short)0, acc3, false, false);
        }
    }
    if constexpr (DUAL) { acc0 = acc0 + acc2; acc1 = acc1 + acc3; }

    #pragma unroll
    for (int half = 0; half < 2; ++half) {
        const int n = (half == 0) ? col0 : col1;
        const v8f& acc = (half == 0) ? acc0 : acc1;
        if (n < nValid) {
            const float bv = bias ? bias[n] : 0.0f;
            #pragma unroll
            for (int r = 0; r < 8; ++r) {
                int m = tm * 16 + r + 8 * hi;
                if (m < mValid) {
                    float val = acc[r] + bv;
                    if (doLeaky) val = (val > 0.0f) ? val : 0.2f * val;
                    if (D32) D32[(size_t)m * ldd + n] = val;
                    if (D16) D16[(size_t)m * ldd + n] = (_Float16)val;
                }
            }
        }
    }
}

// ---------------------------------------------------------------------------
// Support kernels
// ---------------------------------------------------------------------------
__global__ void zero_f32_kernel(float* p, int n) {
    int i = blockIdx.x * blockDim.x + threadIdx.x;
    if (i < n) p[i] = 0.0f;
}

// pooled[b][ch] = max over 14x14 window; stored f16 (A operand of final GEMM).
// 196 floats = 49 aligned float4 loads per (b,ch).
__global__ void maxpool_kernel(const float* __restrict__ f, _Float16* __restrict__ out, int n) {
    int i = blockIdx.x * blockDim.x + threadIdx.x;
    if (i >= n) return;
    const float4* p = (const float4*)(f + (size_t)i * HW2);
    float4 v0 = p[0];
    float m = fmaxf(fmaxf(v0.x, v0.y), fmaxf(v0.z, v0.w));
    #pragma unroll 7
    for (int j = 1; j < HW2 / 4; ++j) {
        float4 v = p[j];
        m = fmaxf(m, fmaxf(fmaxf(v.x, v.y), fmaxf(v.z, v.w)));
    }
    out[i] = (_Float16)m;
}

// dense [rValid][srcCols] f32 -> padded f16 [rows][ldk], zeros outside valid range
__global__ void cvt_pad_kernel(const float* __restrict__ src, _Float16* __restrict__ dst,
                               int rows, int srcCols, int ldk, int rValid, int cValid) {
    int i = blockIdx.x * blockDim.x + threadIdx.x;
    if (i >= rows * ldk) return;
    int r = i / ldk, c = i % ldk;
    float v = (r < rValid && c < cValid) ? src[(size_t)r * srcCols + c] : 0.0f;
    dst[i] = (_Float16)v;
}

// W[K][N] f32 -> Wt[N][ldk] f16 (transposed, K-padded with zeros)
__global__ void cvt_wT_kernel(const float* __restrict__ W, _Float16* __restrict__ Wt,
                              int Ksrc, int N, int ldk) {
    int i = blockIdx.x * blockDim.x + threadIdx.x;
    if (i >= N * ldk) return;
    int n = i / ldk, k = i % ldk;
    float v = (k < Ksrc) ? W[(size_t)k * N + n] : 0.0f;
    Wt[i] = (_Float16)v;
}

// in-degree counts (same for both layers)
__global__ void count_kernel(const int* __restrict__ ei, float* __restrict__ cnt, int E) {
    int e = blockIdx.x * blockDim.x + threadIdx.x;
    if (e >= E) return;
    atomicAdd(&cnt[ei[E + e]], 1.0f);
}

// scatter-add layer 1: ssum[dst][k] += x[src][k]  (x f32, DIN dims)
__global__ void scatter_f32_kernel(const int* __restrict__ ei, const float* __restrict__ x,
                                   float* __restrict__ ssum, int E, int d, int ldk) {
    int i = blockIdx.x * blockDim.x + threadIdx.x;
    if (i >= E * d) return;
    int e = i / d, k = i % d;
    atomicAdd(&ssum[(size_t)ei[E + e] * ldk + k], x[(size_t)ei[e] * d + k]);
}

// scatter-add layer 2: ssum[dst][k] += h1[src][k]  (h1 f16)
__global__ void scatter_f16_kernel(const int* __restrict__ ei, const _Float16* __restrict__ h,
                                   float* __restrict__ ssum, int E, int d) {
    int i = blockIdx.x * blockDim.x + threadIdx.x;
    if (i >= E * d) return;
    int e = i / d, k = i % d;
    atomicAdd(&ssum[(size_t)ei[E + e] * d + k], (float)h[(size_t)ei[e] * d + k]);
}

// mean = ssum / max(cnt,1), converted to f16 padded layout
__global__ void mean_kernel(const float* __restrict__ ssum, const float* __restrict__ cnt,
                            _Float16* __restrict__ out, int rows, int ldk) {
    int i = blockIdx.x * blockDim.x + threadIdx.x;
    if (i >= rows * ldk) return;
    float c = cnt[i / ldk];
    c = (c > 1.0f) ? c : 1.0f;
    out[i] = (_Float16)(ssum[i] / c);
}

// ---------------------------------------------------------------------------
// Host launcher
// ---------------------------------------------------------------------------
extern "C" void kernel_launch(void* const* d_in, const int* in_sizes, int n_in,
                              void* d_out, int out_size, void* d_ws, size_t ws_size,
                              hipStream_t stream) {
    const float* feature = (const float*)d_in[0];   // [64,2048,14,14]
    const float* inp     = (const float*)d_in[1];   // [1,1000,300]
    const int*   ei      = (const int*)  d_in[2];   // [2,40000]
    const float* W_l1    = (const float*)d_in[3];   // [300,1024]
    const float* b_l1    = (const float*)d_in[4];   // [1024]
    const float* W_r1    = (const float*)d_in[5];   // [300,1024]
    const float* W_l2    = (const float*)d_in[6];   // [1024,2048]
    const float* b_l2    = (const float*)d_in[7];   // [2048]
    const float* W_r2    = (const float*)d_in[8];   // [1024,2048]
    float* out = (float*)d_out;                     // [64,1000]

    // Workspace layout (256B aligned regions)
    char* base = (char*)d_ws;
    size_t off = 0;
    auto take = [&](size_t bytes) { size_t r = off; off += (bytes + 255) & ~(size_t)255; return r; };

    _Float16* pooled_h = (_Float16*)(base + take((size_t)Bb * CHn * 2));      // [64][2048]
    _Float16* x_h      = (_Float16*)(base + take((size_t)Cpad * DINp * 2));   // [1024][320]
    _Float16* mean1_h  = (_Float16*)(base + take((size_t)Cpad * DINp * 2));
    _Float16* Wl1_t    = (_Float16*)(base + take((size_t)DH1 * DINp * 2));    // [1024][320]
    _Float16* Wr1_t    = (_Float16*)(base + take((size_t)DH1 * DINp * 2));
    _Float16* h1_h     = (_Float16*)(base + take((size_t)Cpad * DH1 * 2));    // [1024][1024]
    _Float16* mean2_h  = (_Float16*)(base + take((size_t)Cpad * DH1 * 2));
    _Float16* Wl2_t    = (_Float16*)(base + take((size_t)DH2 * DH1 * 2));     // [2048][1024]
    _Float16* Wr2_t    = (_Float16*)(base + take((size_t)DH2 * DH1 * 2));
    _Float16* h2_h     = (_Float16*)(base + take((size_t)Cpad * DH2 * 2));    // [1024][2048]
    // contiguous f32 region that must be zeroed each call:
    size_t zoff = off;
    float* cnt   = (float*)(base + take((size_t)Cpad * 4));
    float* ssum1 = (float*)(base + take((size_t)Cpad * DINp * 4));
    float* ssum2 = (float*)(base + take((size_t)Cpad * DH1 * 4));
    int zeroN = (int)((off - zoff) / 4);
    (void)ws_size; (void)n_in; (void)in_sizes; (void)out_size;

    const int T = 256;
    auto blk = [&](int n) { return (n + T - 1) / T; };

    // 0) zero accumulators (cnt, ssum1, ssum2 are contiguous)
    zero_f32_kernel<<<blk(zeroN), T, 0, stream>>>((float*)(base + zoff), zeroN);

    // 1) max-pool backbone features -> pooled (f16)
    maxpool_kernel<<<blk(Bb * CHn), T, 0, stream>>>(feature, pooled_h, Bb * CHn);

    // 2) stage f16 operands
    cvt_pad_kernel<<<blk(Cpad * DINp), T, 0, stream>>>(inp, x_h, Cpad, DIN, DINp, Cn, DIN);
    cvt_wT_kernel<<<blk(DH1 * DINp), T, 0, stream>>>(W_l1, Wl1_t, DIN, DH1, DINp);
    cvt_wT_kernel<<<blk(DH1 * DINp), T, 0, stream>>>(W_r1, Wr1_t, DIN, DH1, DINp);
    cvt_wT_kernel<<<blk(DH2 * DH1), T, 0, stream>>>(W_l2, Wl2_t, DH1, DH2, DH1);
    cvt_wT_kernel<<<blk(DH2 * DH1), T, 0, stream>>>(W_r2, Wr2_t, DH1, DH2, DH1);

    // 3) graph degree counts (shared by both layers)
    count_kernel<<<blk(En), T, 0, stream>>>(ei, cnt, En);

    // 4) layer-1 scatter mean
    scatter_f32_kernel<<<blk(En * DIN), T, 0, stream>>>(ei, inp, ssum1, En, DIN, DINp);
    mean_kernel<<<blk(Cpad * DINp), T, 0, stream>>>(ssum1, cnt, mean1_h, Cpad, DINp);

    // 5) layer-1 GEMM: h1 = leaky(mean1@Wl1 + x@Wr1 + b1)   [1008 x 1024]
    gemm16x32_wmma_kernel<true><<<dim3(DH1 / 32, 63), dim3(32), 0, stream>>>(
        mean1_h, Wl1_t, x_h, Wr1_t, b_l1, nullptr, h1_h,
        DINp, DINp, DH1, 1008, DH1, 1);

    // 6) layer-2 scatter mean (from h1)
    scatter_f16_kernel<<<blk(En * DH1), T, 0, stream>>>(ei, h1_h, ssum2, En, DH1);
    mean_kernel<<<blk(Cpad * DH1), T, 0, stream>>>(ssum2, cnt, mean2_h, Cpad, DH1);

    // 7) layer-2 GEMM: h2 = mean2@Wl2 + h1@Wr2 + b2        [1008 x 2048]
    gemm16x32_wmma_kernel<true><<<dim3(DH2 / 32, 63), dim3(32), 0, stream>>>(
        mean2_h, Wl2_t, h1_h, Wr2_t, b_l2, nullptr, h2_h,
        DH1, DH1, DH2, 1008, DH2, 0);

    // 8) final GEMM: out = pooled @ h2^T  (B operand = h2 rows, K contiguous).
    //    32 column-pair tiles cover cols 0..1023 (guard stores at 1000).
    gemm16x32_wmma_kernel<false><<<dim3(32, Bb / 16), dim3(32), 0, stream>>>(
        pooled_h, h2_h, nullptr, nullptr, nullptr, out, nullptr,
        DH2, DH2, Cn, Bb, Cn, 0);
}
